// Block_47571057771138
// MI455X (gfx1250) — matrix operand order repaired and checked by reading, not compiled, not run on hardware
//
#include <hip/hip_runtime.h>
#include <hip/hip_bf16.h>

// Transformer block (GPT-2 style) for MI455X / gfx1250.
// B=4, T=2048, C=1024, H=16, D=64, FC=4096.
// All matmuls via v_wmma_f32_16x16x32_f16 (f16 in, f32 accumulate).
//
// Layout decisions (each fragment is one 32B load or two 16B loads):
//  * weights repacked to [K/32][N][32] -> B fragment = 1x h16 load
//  * V stored transposed [channel][token] by its projection GEMM
//  * attention computes S^T = K*Q^T so the score fragment IS the
//    A-operand layout for O += P*V (no LDS transpose needed)
// Both the GEMM k-loop and the attention key-loop are software-pipelined:
// next-step fragments load into fresh registers while the current WMMAs
// (and the softmax VALU chain) execute.

#define C_DIM  1024
#define T_DIM  2048
#define B_DIM  4
#define H_DIM  16
#define HD_DIM 64
#define M_DIM  (B_DIM * T_DIM)   // 8192 token rows
#define FC_DIM 4096

typedef _Float16 h8  __attribute__((ext_vector_type(8)));
typedef _Float16 h16 __attribute__((ext_vector_type(16)));
typedef float    f8  __attribute__((ext_vector_type(8)));

__device__ __forceinline__ f8 wmma_f16(h16 a, h16 b, f8 c) {
  // (neg_a, A, neg_b, B, c_mod, C, reuse_a, reuse_b)
  return __builtin_amdgcn_wmma_f32_16x16x32_f16(false, a, false, b, (short)0, c,
                                                false, false);
}

__device__ __forceinline__ h16 cat8(h8 lo, h8 hi) {
  return __builtin_shufflevector(lo, hi, 0, 1, 2, 3, 4, 5, 6, 7, 8, 9, 10, 11,
                                 12, 13, 14, 15);
}

__device__ __forceinline__ float gelu_f(float v) {
  const float c0 = 0.7978845608028654f;  // sqrt(2/pi)
  return 0.5f * v * (1.f + tanhf(c0 * (v + 0.044715f * v * v * v)));
}

// ---------------------------------------------------------------------------
// Weight repack: f32 row-major [K][N] -> f16 packed [K/32][N][32].
// A lane's B fragment (column n, k-chunk) becomes one contiguous h16.
// ---------------------------------------------------------------------------
__global__ void pack_w_f16(const float* __restrict__ src,
                           _Float16* __restrict__ dst, int K, int N) {
  int idx = blockIdx.x * 256 + threadIdx.x;
  if (idx >= K * N) return;
  int k = idx / N;
  int n = idx - k * N;
  dst[((size_t)(k >> 5) * N + n) * 32 + ((k & 7) | ((k & 8) << 1) | ((k & 16) >> 1))] = (_Float16)src[idx];
}

// ---------------------------------------------------------------------------
// LayerNorm over C=1024, output f16.  One block (8 waves) per token row.
// ---------------------------------------------------------------------------
__global__ __launch_bounds__(256) void ln_f16(const float* __restrict__ x,
                                              const float* __restrict__ g,
                                              const float* __restrict__ bta,
                                              _Float16* __restrict__ y) {
  __shared__ float red[2][8];
  const size_t row = blockIdx.x;
  const float* xr = x + row * C_DIM;
  float s = 0.f, s2 = 0.f;
  for (int i = threadIdx.x; i < C_DIM; i += 256) {
    float v = xr[i];
    s += v;
    s2 += v * v;
  }
#pragma unroll
  for (int o = 1; o < 32; o <<= 1) {
    s  += __shfl_xor(s, o, 32);
    s2 += __shfl_xor(s2, o, 32);
  }
  if ((threadIdx.x & 31) == 0) {
    red[0][threadIdx.x >> 5] = s;
    red[1][threadIdx.x >> 5] = s2;
  }
  __syncthreads();
  s = 0.f; s2 = 0.f;
#pragma unroll
  for (int w = 0; w < 8; ++w) { s += red[0][w]; s2 += red[1][w]; }
  const float mu   = s * (1.f / C_DIM);
  const float var  = s2 * (1.f / C_DIM) - mu * mu;
  const float rstd = rsqrtf(var + 1e-5f);
  _Float16* yr = y + row * C_DIM;
  for (int i = threadIdx.x; i < C_DIM; i += 256)
    yr[i] = (_Float16)((xr[i] - mu) * rstd * g[i] + bta[i]);
}

// ---------------------------------------------------------------------------
// WMMA GEMM:  C[M,N] = act(A[M,K] * B[K,N] + bias) (+ resid)
//   A row-major f16; Bp packed [K/32][N][32] f16.
//   Grid (N/64, M/256); 8 waves/block; each wave owns 32(M) x 64(N).
//   Software-pipelined: k-chunk i+1 fragments load while chunk i's 8 WMMAs
//   issue (distinct registers -> partial loadcnt waits, real overlap).
// EPI: 0 bias->f16 ; 1 bias+resid->f32 ; 2 bias+gelu->f16 ;
//      3 bias->f16 stored transposed [N][M] (for V)
// ---------------------------------------------------------------------------
template <int EPI>
__global__ __launch_bounds__(256) void gemm_wmma(
    const _Float16* __restrict__ A, const _Float16* __restrict__ Bp,
    const float* __restrict__ bias, const float* __restrict__ resid,
    _Float16* __restrict__ Oh, float* __restrict__ Of, int M, int N, int K) {
  const int wave = threadIdx.x >> 5;
  const int lane = threadIdx.x & 31;
  const int half = lane >> 4;
  const int ln16 = lane & 15;
  const int n0 = blockIdx.x * 64;
  const int m0 = blockIdx.y * 256 + wave * 32;
  const int abase = half * 8;
  const int bbase = half * 16;

  f8 acc[2][4] = {};
  const _Float16* arow0 = A + (size_t)(m0 + ln16) * K;
  const _Float16* arow1 = A + (size_t)(m0 + 16 + ln16) * K;
  const _Float16* bcol = Bp + ((size_t)(n0 + ln16)) * 32 + bbase;

  // prologue: fragments for k-chunk 0
  h16 a0 = cat8(*(const h8*)(arow0 + abase), *(const h8*)(arow0 + 16 + abase));
  h16 a1 = cat8(*(const h8*)(arow1 + abase), *(const h8*)(arow1 + 16 + abase));
  h16 bf0 = *(const h16*)(bcol);
  h16 bf1 = *(const h16*)(bcol + 512);
  h16 bf2 = *(const h16*)(bcol + 1024);
  h16 bf3 = *(const h16*)(bcol + 1536);

  for (int k0 = 0; k0 < K; k0 += 32) {
    // issue next chunk's loads first (last iter harmlessly reloads chunk 0)
    const int k1 = (k0 + 32 < K) ? (k0 + 32) : 0;
    const h16 na0 = cat8(*(const h8*)(arow0 + k1 + abase),
                         *(const h8*)(arow0 + k1 + 16 + abase));
    const h16 na1 = cat8(*(const h8*)(arow1 + k1 + abase),
                         *(const h8*)(arow1 + k1 + 16 + abase));
    const _Float16* nb = bcol + (size_t)(k1 >> 5) * N * 32;
    __builtin_prefetch(nb + (size_t)N * 32, 0, 0);
    const h16 nbf0 = *(const h16*)(nb);
    const h16 nbf1 = *(const h16*)(nb + 512);
    const h16 nbf2 = *(const h16*)(nb + 1024);
    const h16 nbf3 = *(const h16*)(nb + 1536);

    acc[0][0] = wmma_f16(a0, bf0, acc[0][0]);
    acc[1][0] = wmma_f16(a1, bf0, acc[1][0]);
    acc[0][1] = wmma_f16(a0, bf1, acc[0][1]);
    acc[1][1] = wmma_f16(a1, bf1, acc[1][1]);
    acc[0][2] = wmma_f16(a0, bf2, acc[0][2]);
    acc[1][2] = wmma_f16(a1, bf2, acc[1][2]);
    acc[0][3] = wmma_f16(a0, bf3, acc[0][3]);
    acc[1][3] = wmma_f16(a1, bf3, acc[1][3]);

    a0 = na0; a1 = na1;
    bf0 = nbf0; bf1 = nbf1; bf2 = nbf2; bf3 = nbf3;
  }

  // Epilogue: C layout -> M = r + 8*half, N = ln16
#pragma unroll
  for (int mt = 0; mt < 2; ++mt) {
#pragma unroll
    for (int nt = 0; nt < 4; ++nt) {
      const int gn = n0 + nt * 16 + ln16;
      const float bv = bias[gn];
      if (EPI == 3) {  // transposed f16 store: 8 consecutive tokens -> one b128
        h8 st;
#pragma unroll
        for (int r = 0; r < 8; ++r) st[r] = (_Float16)(acc[mt][nt][r] + bv);
        *(h8*)(Oh + (size_t)gn * M + m0 + mt * 16 + 8 * half) = st;
      } else {
#pragma unroll
        for (int r = 0; r < 8; ++r) {
          const int gm = m0 + mt * 16 + r + 8 * half;
          float v = acc[mt][nt][r] + bv;
          if (EPI == 2) v = gelu_f(v);
          if (EPI == 1) {
            v += resid[(size_t)gm * N + gn];
            Of[(size_t)gm * N + gn] = v;
          } else {
            Oh[(size_t)gm * N + gn] = (_Float16)v;
          }
        }
      }
    }
  }
}

// ---------------------------------------------------------------------------
// Flash attention, causal.  One wave per 16-query block per (b,h).
//   S^T = K*Q^T (keys = M dim): the resulting C fragment layout equals the
//   A-operand layout of P for O += P*V, so P never leaves registers.
//   Software-pipelined: the 4 K A-fragments for step kb are loaded during
//   step kb-32's softmax; V fragments load right after the score WMMAs.
//   Q,K row-major [token][channel]; V transposed [channel][token].
// ---------------------------------------------------------------------------
__global__ __launch_bounds__(256) void attn_wmma(const _Float16* __restrict__ Q,
                                                 const _Float16* __restrict__ K,
                                                 const _Float16* __restrict__ Vt,
                                                 _Float16* __restrict__ O) {
  const int lane = threadIdx.x & 31;
  const int wave = threadIdx.x >> 5;
  const int half = lane >> 4;
  const int ln16 = lane & 15;

  const int gw = blockIdx.x * 8 + wave;      // 8192 waves total
  const int qb = gw & (T_DIM / 16 - 1);
  const int h  = (gw >> 7) & (H_DIM - 1);
  const int b  = gw >> 11;
  const int q0 = qb * 16;

  const _Float16* Qb = Q + ((size_t)b * T_DIM) * C_DIM + h * HD_DIM;
  const _Float16* Kb = K + ((size_t)b * T_DIM) * C_DIM + h * HD_DIM;
  const _Float16* Vh = Vt + (size_t)(h * HD_DIM) * M_DIM + (size_t)b * T_DIM;

  const int abase = half * 16;
  const int bbase = half * 16;

  auto ldA = [&](const _Float16* kr, int kc) -> h16 {  // 16x32 A fragment slice
    return cat8(*(const h8*)(kr + kc + abase),
                *(const h8*)(kr + kc + 8 + abase));
  };

  // Q^T B-fragments (column = query q0+ln16), hoisted out of the k loop
  const _Float16* qrow = Qb + (size_t)(q0 + ln16) * C_DIM;
  const h16 qbf0 = *(const h16*)(qrow + bbase);        // dims 0..31 slice
  const h16 qbf1 = *(const h16*)(qrow + 32 + bbase);   // dims 32..63 slice

  f8 oacc[4] = {};
  float mrow = -3.0e38f, lrow = 0.f;  // stats for this lane's query
  const int myq = q0 + ln16;
  const float sc = 0.125f;            // 1/sqrt(64)

  // prologue: K fragments for kb = 0 (tokens 0..31, always in range)
  h16 ka00, ka01, ka10, ka11;
  {
    const _Float16* k0r = Kb + (size_t)ln16 * C_DIM;
    const _Float16* k1r = Kb + (size_t)(16 + ln16) * C_DIM;
    ka00 = ldA(k0r, 0);
    ka01 = ldA(k0r, 32);
    ka10 = ldA(k1r, 0);
    ka11 = ldA(k1r, 32);
  }

  for (int kb = 0; kb < q0 + 16; kb += 32) {
    // ---- score WMMAs: fragments already resident, no load wait
    f8 st0 = {}, st1 = {};
    st0 = wmma_f16(ka00, qbf0, st0);
    st0 = wmma_f16(ka01, qbf1, st0);
    st1 = wmma_f16(ka10, qbf0, st1);
    st1 = wmma_f16(ka11, qbf1, st1);

    // ---- issue next iteration's K loads (overlap with softmax below);
    //      last iteration harmlessly reloads clamped rows
    int nt0 = kb + 32 + ln16;      if (nt0 > T_DIM - 1) nt0 = T_DIM - 1;
    int nt1 = kb + 48 + ln16;      if (nt1 > T_DIM - 1) nt1 = T_DIM - 1;
    const _Float16* nk0r = Kb + (size_t)nt0 * C_DIM;
    const _Float16* nk1r = Kb + (size_t)nt1 * C_DIM;
    const h16 nka00 = ldA(nk0r, 0);
    const h16 nka01 = ldA(nk0r, 32);
    const h16 nka10 = ldA(nk1r, 0);
    const h16 nka11 = ldA(nk1r, 32);

    // ---- V fragments: independent of softmax, issue loads now
    int t0v = kb + half * 8;
    if (t0v > T_DIM - 16) t0v = T_DIM - 16;  // masked tail, keep in-bounds
    const h16 bvf0 = cat8(*(const h8*)(Vh + (size_t)(0 * 16 + ln16) * M_DIM + t0v), *(const h8*)(Vh + (size_t)(0 * 16 + ln16) * M_DIM + t0v + 16));
    const h16 bvf1 = cat8(*(const h8*)(Vh + (size_t)(1 * 16 + ln16) * M_DIM + t0v), *(const h8*)(Vh + (size_t)(1 * 16 + ln16) * M_DIM + t0v + 16));
    const h16 bvf2 = cat8(*(const h8*)(Vh + (size_t)(2 * 16 + ln16) * M_DIM + t0v), *(const h8*)(Vh + (size_t)(2 * 16 + ln16) * M_DIM + t0v + 16));
    const h16 bvf3 = cat8(*(const h8*)(Vh + (size_t)(3 * 16 + ln16) * M_DIM + t0v), *(const h8*)(Vh + (size_t)(3 * 16 + ln16) * M_DIM + t0v + 16));

    // ---- causal mask + online softmax (one query per lane)
    float v0[8], v1[8];
    float vmax = -3.0e38f;
#pragma unroll
    for (int j = 0; j < 8; ++j) {
      const int kg = kb + j + 8 * half;   // key of st0 reg j
      float a0 = st0[j] * sc;
      float a1 = st1[j] * sc;
      if (kg > myq)      a0 = -3.0e38f;
      if (kg + 16 > myq) a1 = -3.0e38f;
      v0[j] = a0;
      v1[j] = a1;
      vmax = fmaxf(vmax, fmaxf(a0, a1));
    }
    vmax = fmaxf(vmax, __shfl_xor(vmax, 16, 32));  // other half of key range
    const float mnew  = fmaxf(mrow, vmax);
    const float alpha = __expf(mrow - mnew);
    mrow = mnew;

    h16 pa;             // P fragment, already in A-operand layout
    float rs = 0.f;
#pragma unroll
    for (int j = 0; j < 8; ++j) {
      const float e0 = __expf(v0[j] - mnew);
      const float e1 = __expf(v1[j] - mnew);
      rs += e0 + e1;
      pa[j]     = (_Float16)e0;
      pa[8 + j] = (_Float16)e1;
    }
    rs += __shfl_xor(rs, 16, 32);
    lrow = lrow * alpha + rs;

    // broadcast alpha into O-accumulator row layout and rescale
#pragma unroll
    for (int r = 0; r < 8; ++r) {
      const float ar = __shfl(alpha, r + 8 * half, 32);
#pragma unroll
      for (int nt = 0; nt < 4; ++nt) oacc[nt][r] *= ar;
    }

    // ---- O += P * V
    oacc[0] = wmma_f16(pa, bvf0, oacc[0]);
    oacc[1] = wmma_f16(pa, bvf1, oacc[1]);
    oacc[2] = wmma_f16(pa, bvf2, oacc[2]);
    oacc[3] = wmma_f16(pa, bvf3, oacc[3]);

    // rotate pipelined K fragments
    ka00 = nka00; ka01 = nka01; ka10 = nka10; ka11 = nka11;
  }

  // ---- normalize + store (f16, token-major for the next GEMM)
  _Float16* Ob = O + ((size_t)b * T_DIM) * C_DIM + h * HD_DIM;
#pragma unroll
  for (int r = 0; r < 8; ++r) {
    const float lr = __shfl(lrow, r + 8 * half, 32);
    const float invl = 1.f / lr;
    const int t = q0 + r + 8 * half;
#pragma unroll
    for (int nt = 0; nt < 4; ++nt)
      Ob[(size_t)t * C_DIM + nt * 16 + ln16] = (_Float16)(oacc[nt][r] * invl);
  }
}

// ---------------------------------------------------------------------------
// Host-side orchestration
// ---------------------------------------------------------------------------
extern "C" void kernel_launch(void* const* d_in, const int* in_sizes, int n_in,
                              void* d_out, int out_size, void* d_ws,
                              size_t ws_size, hipStream_t stream) {
  (void)in_sizes; (void)n_in; (void)out_size; (void)ws_size;

  const float* x   = (const float*)d_in[0];
  const float* Wq  = (const float*)d_in[1];
  const float* bq  = (const float*)d_in[2];
  const float* Wk  = (const float*)d_in[3];
  const float* bk  = (const float*)d_in[4];
  const float* Wv  = (const float*)d_in[5];
  const float* bv  = (const float*)d_in[6];
  const float* Wo  = (const float*)d_in[7];
  const float* bo  = (const float*)d_in[8];
  const float* g1  = (const float*)d_in[9];
  const float* b1  = (const float*)d_in[10];
  const float* g2  = (const float*)d_in[11];
  const float* b2  = (const float*)d_in[12];
  const float* Wfc = (const float*)d_in[13];
  const float* bfc = (const float*)d_in[14];
  const float* Wpr = (const float*)d_in[15];
  const float* bpr = (const float*)d_in[16];
  float* out = (float*)d_out;

  char* ws = (char*)d_ws;
  size_t off = 0;
  auto alloc = [&](size_t bytes) -> void* {
    size_t cur = off;
    off = (off + bytes + 255) & ~(size_t)255;
    return (void*)(ws + cur);
  };

  const size_t MC = (size_t)M_DIM * C_DIM;
  _Float16* wqp = (_Float16*)alloc((size_t)C_DIM * C_DIM * 2);
  _Float16* wkp = (_Float16*)alloc((size_t)C_DIM * C_DIM * 2);
  _Float16* wvp = (_Float16*)alloc((size_t)C_DIM * C_DIM * 2);
  _Float16* wop = (_Float16*)alloc((size_t)C_DIM * C_DIM * 2);
  _Float16* wfcp = (_Float16*)alloc((size_t)C_DIM * FC_DIM * 2);
  _Float16* wprp = (_Float16*)alloc((size_t)C_DIM * FC_DIM * 2);
  _Float16* ln1h = (_Float16*)alloc(MC * 2);
  _Float16* qh   = (_Float16*)alloc(MC * 2);
  _Float16* kh   = (_Float16*)alloc(MC * 2);
  _Float16* vth  = (_Float16*)alloc(MC * 2);   // transposed [channel][token]
  _Float16* atth = (_Float16*)alloc(MC * 2);
  float*    x2   = (float*)alloc(MC * 4);
  _Float16* ln2h = (_Float16*)alloc(MC * 2);
  _Float16* fch  = (_Float16*)alloc((size_t)M_DIM * FC_DIM * 2);

  // packed f16 weights
  {
    const int nsq = C_DIM * C_DIM;
    const int nfc = C_DIM * FC_DIM;
    pack_w_f16<<<(nsq + 255) / 256, 256, 0, stream>>>(Wq, wqp, C_DIM, C_DIM);
    pack_w_f16<<<(nsq + 255) / 256, 256, 0, stream>>>(Wk, wkp, C_DIM, C_DIM);
    pack_w_f16<<<(nsq + 255) / 256, 256, 0, stream>>>(Wv, wvp, C_DIM, C_DIM);
    pack_w_f16<<<(nsq + 255) / 256, 256, 0, stream>>>(Wo, wop, C_DIM, C_DIM);
    pack_w_f16<<<(nfc + 255) / 256, 256, 0, stream>>>(Wfc, wfcp, C_DIM, FC_DIM);
    pack_w_f16<<<(nfc + 255) / 256, 256, 0, stream>>>(Wpr, wprp, FC_DIM, C_DIM);
  }

  const dim3 blk(256);
  const dim3 gC(C_DIM / 64, M_DIM / 256);    // N=1024 GEMMs
  const dim3 gF(FC_DIM / 64, M_DIM / 256);   // N=4096 GEMM

  // ln1
  ln_f16<<<M_DIM, blk, 0, stream>>>(x, g1, b1, ln1h);
  // q,k,v projections (v stored transposed)
  gemm_wmma<0><<<gC, blk, 0, stream>>>(ln1h, wqp, bq, nullptr, qh, nullptr,
                                       M_DIM, C_DIM, C_DIM);
  gemm_wmma<0><<<gC, blk, 0, stream>>>(ln1h, wkp, bk, nullptr, kh, nullptr,
                                       M_DIM, C_DIM, C_DIM);
  gemm_wmma<3><<<gC, blk, 0, stream>>>(ln1h, wvp, bv, nullptr, vth, nullptr,
                                       M_DIM, C_DIM, C_DIM);
  // causal flash attention
  attn_wmma<<<M_DIM / 16 / 8, blk, 0, stream>>>(qh, kh, vth, atth);
  // output projection + residual -> x2 (f32)
  gemm_wmma<1><<<gC, blk, 0, stream>>>(atth, wop, bo, x, nullptr, x2,
                                       M_DIM, C_DIM, C_DIM);
  // ln2
  ln_f16<<<M_DIM, blk, 0, stream>>>(x2, g2, b2, ln2h);
  // fc + gelu
  gemm_wmma<2><<<gF, blk, 0, stream>>>(ln2h, wfcp, bfc, nullptr, fch, nullptr,
                                       M_DIM, FC_DIM, C_DIM);
  // proj + residual -> out (f32)
  gemm_wmma<1><<<gC, blk, 0, stream>>>(fch, wprp, bpr, x2, nullptr, out,
                                       M_DIM, C_DIM, FC_DIM);
}
